// VP_Refine_Module_gated_self_attn_50586124812575
// MI455X (gfx1250) — matrix operand order repaired and checked by reading, not compiled, not run on hardware
//
#include <hip/hip_runtime.h>
#include <hip/hip_bf16.h>

// ---------------- types ----------------
typedef __bf16 bf16;
typedef __attribute__((ext_vector_type(16))) __bf16 v16bf;
typedef __attribute__((ext_vector_type(8)))  float  v8f;
typedef __attribute__((ext_vector_type(4)))  int    v4i;

#define E_  1024
#define H_  8
#define HD_ 128
#define L_  4
#define B_  64
#define M_  256
#define TOK (B_ * M_)          // 16384 tokens
#define BHM (B_ * H_ * M_)     // 131072

// -------- CDNA5 async global->LDS path (guarded; falls back to sync copies) --------
#if defined(__has_builtin)
#if __has_builtin(__builtin_amdgcn_global_load_async_to_lds_b128)
#define HAVE_ASYNC 1
#endif
#endif
#ifndef HAVE_ASYNC
#define HAVE_ASYNC 0
#endif

#if HAVE_ASYNC
typedef __attribute__((address_space(1))) v4i* gv4i_p;
typedef __attribute__((address_space(3))) v4i* lv4i_p;
#endif

static __device__ __forceinline__ void cp16(bf16* dst, const bf16* src) {
#if HAVE_ASYNC
    __builtin_amdgcn_global_load_async_to_lds_b128(
        (gv4i_p)(unsigned long long)(const void*)src,
        (lv4i_p)(unsigned int)(unsigned long long)(const void*)dst, 0, 0);
#else
    *(int4*)dst = *(const int4*)src;
#endif
}

static __device__ __forceinline__ void wait_async() {
#if HAVE_ASYNC
#if __has_builtin(__builtin_amdgcn_s_wait_asynccnt)
    __builtin_amdgcn_s_wait_asynccnt(0);
#else
    asm volatile("s_wait_asynccnt 0x0" ::: "memory");
#endif
#endif
}

static __device__ __forceinline__ v8f vzero8() {
    v8f z = {0.f, 0.f, 0.f, 0.f, 0.f, 0.f, 0.f, 0.f};
    return z;
}

static __device__ __forceinline__ v8f wmma_bf16(v16bf a, v16bf b, v8f c) {
    return __builtin_amdgcn_wmma_f32_16x16x32_bf16(false, a, false, b, (short)0, c,
                                                   false, false);
}

// A fragment: 16x32 bf16 (ISA 7.12.2 layout)
static __device__ __forceinline__ v16bf frag_a(const bf16* base, int ld, int lane) {
    int row = lane & 15;
    int kb  = (lane >> 4) << 3;
    const bf16* p = base + row * ld + kb;
    union { v16bf v; int4 q[2]; } u;
    u.q[0] = *(const int4*)(p);
    u.q[1] = *(const int4*)(p + 16);
    return u.v;
}

// B fragment: 32x16 bf16 from an [n][k] tile (contiguous K per output column)
static __device__ __forceinline__ v16bf frag_b(const bf16* base, int ld, int lane) {
    int n  = lane & 15;
    int kb = (lane >> 4) << 3;
    const bf16* p = base + n * ld + kb;
    union { v16bf v; int4 q[2]; } u;
    u.q[0] = *(const int4*)(p);
    u.q[1] = *(const int4*)(p + 16);
    return u.v;
}

// copy rows x cols (cols % 8 == 0) of bf16 global -> LDS in 16B chunks (async if available)
static __device__ __forceinline__ void stage_tile(bf16* __restrict__ dst, int dld,
                                                  const bf16* __restrict__ src, long sld,
                                                  int rows, int cols, int tid, int nthr) {
    int cpr = cols >> 3;
    int chunks = rows * cpr;
    for (int c = tid; c < chunks; c += nthr) {
        int r  = c / cpr;
        int cc = (c - r * cpr) << 3;
        cp16(dst + r * dld + cc, src + (long)r * sld + cc);
    }
}

// ---------------- 128x128-tile bf16 GEMM (flop-dominant path), double-buffered async ----------------
// A: [rows][K] (lda), Bw: [N][K] (ldb), out f32 (ldo). mode 1 => celu epilogue.
#define G128_SMEM (4 * 128 * 72 * 2)
__global__ __launch_bounds__(256)
void gemm128_kernel(const bf16* __restrict__ A, long lda,
                    const bf16* __restrict__ Bw, long ldb,
                    const float* __restrict__ bias,
                    float* __restrict__ out, long ldo, int K, int mode) {
    extern __shared__ char sm[];
    bf16* AsB = (bf16*)sm;              // 2 x [128][72]
    bf16* BsB = AsB + 2 * 128 * 72;     // 2 x [128][72]
    const int tid = threadIdx.x, lane = tid & 31, wave = tid >> 5;
    const int row0 = blockIdx.x * 128;
    const int col0 = blockIdx.y * 128;

    v8f acc[8];
#pragma unroll
    for (int j = 0; j < 8; ++j) acc[j] = vzero8();

    stage_tile(AsB, 72, A + (long)row0 * lda, lda, 128, 64, tid, 256);
    stage_tile(BsB, 72, Bw + (long)col0 * ldb, ldb, 128, 64, tid, 256);
    wait_async();
    __syncthreads();

    for (int k0 = 0; k0 < K; k0 += 64) {
        const int cur = (k0 >> 6) & 1;
        bf16* As = AsB + cur * 128 * 72;
        bf16* Bs = BsB + cur * 128 * 72;
        if (k0 + 64 < K) {  // prefetch next K-slab into the other buffer
            bf16* An = AsB + (cur ^ 1) * 128 * 72;
            bf16* Bn = BsB + (cur ^ 1) * 128 * 72;
            stage_tile(An, 72, A + (long)row0 * lda + k0 + 64, lda, 128, 64, tid, 256);
            stage_tile(Bn, 72, Bw + (long)col0 * ldb + k0 + 64, ldb, 128, 64, tid, 256);
        }
#pragma unroll
        for (int kk = 0; kk < 64; kk += 32) {
            v16bf a = frag_a(As + wave * 16 * 72 + kk, 72, lane);
#pragma unroll
            for (int g = 0; g < 2; ++g) {
                v16bf b0 = frag_b(Bs + (g * 4 + 0) * 16 * 72 + kk, 72, lane);
                v16bf b1 = frag_b(Bs + (g * 4 + 1) * 16 * 72 + kk, 72, lane);
                v16bf b2 = frag_b(Bs + (g * 4 + 2) * 16 * 72 + kk, 72, lane);
                v16bf b3 = frag_b(Bs + (g * 4 + 3) * 16 * 72 + kk, 72, lane);
                acc[g * 4 + 0] = wmma_bf16(a, b0, acc[g * 4 + 0]);
                acc[g * 4 + 1] = wmma_bf16(a, b1, acc[g * 4 + 1]);
                acc[g * 4 + 2] = wmma_bf16(a, b2, acc[g * 4 + 2]);
                acc[g * 4 + 3] = wmma_bf16(a, b3, acc[g * 4 + 3]);
            }
        }
        wait_async();
        __syncthreads();
    }
    const int rbase = row0 + wave * 16 + ((lane >> 4) << 3);
#pragma unroll
    for (int j = 0; j < 8; ++j) {
        int col  = col0 + j * 16 + (lane & 15);
        float bv = bias ? bias[col] : 0.f;
#pragma unroll
        for (int r = 0; r < 8; ++r) {
            float v = acc[j][r] + bv;
            if (mode == 1) v = (v > 0.f) ? v : 1.3f * expm1f(v * (1.0f / 1.3f));
            out[(long)(rbase + r) * ldo + col] = v;
        }
    }
}

// ---------------- 64x128-tile bf16 GEMM (small-M path), double-buffered async ----------------
__global__ __launch_bounds__(256)
void gemm_bf16_kernel(const bf16* __restrict__ A, long lda,
                      const bf16* __restrict__ Bw, long ldb,
                      const float* __restrict__ bias,
                      float* __restrict__ out, long ldo, int K, int mode) {
    __shared__ bf16 As[2][64 * 72];
    __shared__ bf16 Bs[2][128 * 72];
    const int tid  = threadIdx.x;
    const int lane = tid & 31, wave = tid >> 5;
    const int row0 = blockIdx.x * 64;
    const int col0 = blockIdx.y * 128;
    const int rt = wave >> 1;
    const int cg = wave & 1;

    v8f acc[4];
#pragma unroll
    for (int j = 0; j < 4; ++j) acc[j] = vzero8();

    stage_tile(As[0], 72, A + (long)row0 * lda, lda, 64, 64, tid, 256);
    stage_tile(Bs[0], 72, Bw + (long)col0 * ldb, ldb, 128, 64, tid, 256);
    wait_async();
    __syncthreads();

    for (int k0 = 0; k0 < K; k0 += 64) {
        const int cur = (k0 >> 6) & 1;
        if (k0 + 64 < K) {
            stage_tile(As[cur ^ 1], 72, A + (long)row0 * lda + k0 + 64, lda, 64, 64, tid, 256);
            stage_tile(Bs[cur ^ 1], 72, Bw + (long)col0 * ldb + k0 + 64, ldb, 128, 64, tid, 256);
        }
#pragma unroll
        for (int kk = 0; kk < 64; kk += 32) {
            v16bf a  = frag_a(As[cur] + rt * 16 * 72 + kk, 72, lane);
            v16bf b0 = frag_b(Bs[cur] + (cg * 64 + 0)  * 72 + kk, 72, lane);
            v16bf b1 = frag_b(Bs[cur] + (cg * 64 + 16) * 72 + kk, 72, lane);
            v16bf b2 = frag_b(Bs[cur] + (cg * 64 + 32) * 72 + kk, 72, lane);
            v16bf b3 = frag_b(Bs[cur] + (cg * 64 + 48) * 72 + kk, 72, lane);
            acc[0] = wmma_bf16(a, b0, acc[0]);
            acc[1] = wmma_bf16(a, b1, acc[1]);
            acc[2] = wmma_bf16(a, b2, acc[2]);
            acc[3] = wmma_bf16(a, b3, acc[3]);
        }
        wait_async();
        __syncthreads();
    }
    const int rbase = row0 + rt * 16 + ((lane >> 4) << 3);
#pragma unroll
    for (int j = 0; j < 4; ++j) {
        int col  = col0 + cg * 64 + j * 16 + (lane & 15);
        float bv = bias ? bias[col] : 0.f;
#pragma unroll
        for (int r = 0; r < 8; ++r) {
            float v = acc[j][r] + bv;
            if (mode == 1) v = (v > 0.f) ? v : 1.3f * expm1f(v * (1.0f / 1.3f));
            out[(long)(rbase + r) * ldo + col] = v;
        }
    }
}

// ---------------- group norm: one wave per (token, head) group of 128 ----------------
__global__ __launch_bounds__(256)
void groupnorm_kernel(const float* __restrict__ src, const float* __restrict__ w,
                      const float* __restrict__ bb, bf16* __restrict__ dst) {
    int wave = threadIdx.x >> 5, lane = threadIdx.x & 31;
    long g = (long)blockIdx.x * 8 + wave;
    long token = g >> 3;
    int  h = (int)(g & 7);
    const float* p = src + token * E_ + h * HD_;
    float4 v = *(const float4*)(p + lane * 4);
    float s  = v.x + v.y + v.z + v.w;
    float s2 = v.x * v.x + v.y * v.y + v.z * v.z + v.w * v.w;
#pragma unroll
    for (int m = 16; m; m >>= 1) {
        s  += __shfl_xor(s, m, 32);
        s2 += __shfl_xor(s2, m, 32);
    }
    float mu = s * (1.f / HD_);
    float rs = rsqrtf(s2 * (1.f / HD_) - mu * mu + 1e-5f);
    int e0 = h * HD_ + lane * 4;
    bf16* q = dst + token * E_ + e0;
    q[0] = (bf16)((v.x - mu) * rs * w[e0 + 0] + bb[e0 + 0]);
    q[1] = (bf16)((v.y - mu) * rs * w[e0 + 1] + bb[e0 + 1]);
    q[2] = (bf16)((v.z - mu) * rs * w[e0 + 2] + bb[e0 + 2]);
    q[3] = (bf16)((v.w - mu) * rs * w[e0 + 3] + bb[e0 + 3]);
}

// ---------------- fused attention per (b,h, 64-query tile) ----------------
// smem: Sq f32[64][256] | P bf16[64][256] | Xq bf16[64][136] | Kst bf16[64][136] | Vst bf16[128][72]
#define ATTN_SMEM (64 * 256 * 4 + 64 * 256 * 2 + 2 * 64 * 136 * 2 + 128 * 72 * 2)
__global__ __launch_bounds__(256)
void attn_kernel(const bf16* __restrict__ xb, const bf16* __restrict__ memb,
                 const bf16* __restrict__ xT, const float* __restrict__ mask,
                 bf16* __restrict__ inp) {
    extern __shared__ char smem[];
    float* Sq  = (float*)smem;                                   // [64][256]
    bf16*  P   = (bf16*)(smem + 64 * 256 * 4);                   // [64][256]
    bf16*  Xq  = (bf16*)(smem + 64 * 256 * 4 + 64 * 256 * 2);    // [64][136]
    bf16*  Kst = Xq + 64 * 136;                                  // [64][136]
    bf16*  Vst = Kst + 64 * 136;                                 // [128][72]

    const int tid  = threadIdx.x;
    const int lane = tid & 31, wave = tid >> 5;
    const int bh = blockIdx.x;        // 0..511
    const int b  = bh >> 3, h = bh & 7;
    const int q0 = blockIdx.y * 64;

    // stage queries (and write xh half of inp)
    stage_tile(Xq, 136, xb + ((long)(b * M_ + q0)) * E_ + h * HD_, E_, 64, 128, tid, 256);
    wait_async();
    __syncthreads();
    for (int c = tid; c < 64 * 16; c += 256) {
        int r = c >> 4, cc = (c & 15) << 3;
        *(int4*)(inp + ((long)bh * M_ + q0 + r) * 256 + cc) = *(const int4*)(Xq + r * 136 + cc);
    }

    // ---- scores S = Xq @ Mh^T * scale ----
    const float scal = 0.08838834764831845f; // 128^-0.5
    for (int kb = 0; kb < 4; ++kb) {
        __syncthreads();
        stage_tile(Kst, 136, memb + ((long)(b * M_ + kb * 64)) * E_ + h * HD_, E_, 64, 128, tid, 256);
        wait_async();
        __syncthreads();
        int rt = wave >> 1;
        int c0t = (wave & 1);
        v8f a0 = vzero8(), a1 = vzero8();
#pragma unroll
        for (int kk = 0; kk < 128; kk += 32) {
            v16bf a  = frag_a(Xq + rt * 16 * 136 + kk, 136, lane);
            v16bf b0 = frag_b(Kst + (c0t * 16) * 136 + kk, 136, lane);
            v16bf b1 = frag_b(Kst + ((c0t + 2) * 16) * 136 + kk, 136, lane);
            a0 = wmma_bf16(a, b0, a0);
            a1 = wmma_bf16(a, b1, a1);
        }
        int rbase = rt * 16 + ((lane >> 4) << 3);
        int cc0 = kb * 64 + c0t * 16 + (lane & 15);
#pragma unroll
        for (int r = 0; r < 8; ++r) {
            Sq[(rbase + r) * 256 + cc0]      = a0[r] * scal;
            Sq[(rbase + r) * 256 + cc0 + 32] = a1[r] * scal;
        }
    }
    __syncthreads();

    // ---- masked softmax, write bf16 P ----
    for (int r = wave; r < 64; r += 8) {
        float vals[8];
        float mx = -3.0e38f;
#pragma unroll
        for (int j = 0; j < 8; ++j) {
            int c = lane + 32 * j;
            float mk = mask[b * M_ + c];
            float v = (mk == 0.f) ? -1.0e9f : Sq[r * 256 + c];
            vals[j] = v;
            mx = fmaxf(mx, v);
        }
#pragma unroll
        for (int m = 16; m; m >>= 1) mx = fmaxf(mx, __shfl_xor(mx, m, 32));
        float sum = 0.f;
#pragma unroll
        for (int j = 0; j < 8; ++j) { vals[j] = expf(vals[j] - mx); sum += vals[j]; }
#pragma unroll
        for (int m = 16; m; m >>= 1) sum += __shfl_xor(sum, m, 32);
        float inv = 1.f / sum;
#pragma unroll
        for (int j = 0; j < 8; ++j) P[r * 256 + lane + 32 * j] = (bf16)(vals[j] * inv);
    }

    // ---- ctx = P @ Xh (values from head-transposed layout) ----
    const int rt = wave >> 1, cg = wave & 1;
    v8f acc[4];
#pragma unroll
    for (int j = 0; j < 4; ++j) acc[j] = vzero8();
    for (int kb = 0; kb < 4; ++kb) {
        __syncthreads();
        stage_tile(Vst, 72, xT + ((long)bh * HD_) * M_ + kb * 64, M_, 128, 64, tid, 256);
        wait_async();
        __syncthreads();
#pragma unroll
        for (int kk = 0; kk < 64; kk += 32) {
            v16bf a  = frag_a(P + rt * 16 * 256 + kb * 64 + kk, 256, lane);
            v16bf b0 = frag_b(Vst + (cg * 64 + 0)  * 72 + kk, 72, lane);
            v16bf b1 = frag_b(Vst + (cg * 64 + 16) * 72 + kk, 72, lane);
            v16bf b2 = frag_b(Vst + (cg * 64 + 32) * 72 + kk, 72, lane);
            v16bf b3 = frag_b(Vst + (cg * 64 + 48) * 72 + kk, 72, lane);
            acc[0] = wmma_bf16(a, b0, acc[0]);
            acc[1] = wmma_bf16(a, b1, acc[1]);
            acc[2] = wmma_bf16(a, b2, acc[2]);
            acc[3] = wmma_bf16(a, b3, acc[3]);
        }
    }
    int rbase = rt * 16 + ((lane >> 4) << 3);
#pragma unroll
    for (int j = 0; j < 4; ++j) {
        int d = cg * 64 + j * 16 + (lane & 15);
#pragma unroll
        for (int r = 0; r < 8; ++r)
            inp[((long)bh * M_ + q0 + rbase + r) * 256 + 128 + d] = (bf16)(acc[j][r]);
    }
}

// ---------------- fused upd/gate dual GEMM + gating ----------------
__global__ __launch_bounds__(256)
void updgate_kernel(const bf16* __restrict__ inp, const bf16* __restrict__ Wu,
                    const bf16* __restrict__ Wg, const float* __restrict__ bu,
                    const float* __restrict__ bg, const float* __restrict__ xold,
                    float* __restrict__ xnew) {
    __shared__ bf16 As[64 * 72];
    __shared__ bf16 Bu[128 * 72];
    __shared__ bf16 Bg[128 * 72];
    const int tid  = threadIdx.x;
    const int lane = tid & 31, wave = tid >> 5;
    const int row0 = blockIdx.x * 64;   // token index in [0, BHM)
    const int rt = wave >> 1, cg = wave & 1;

    v8f accU[4], accG[4];
#pragma unroll
    for (int j = 0; j < 4; ++j) { accU[j] = vzero8(); accG[j] = vzero8(); }

    for (int k0 = 0; k0 < 256; k0 += 64) {
        __syncthreads();
        stage_tile(As, 72, inp + (long)row0 * 256 + k0, 256, 64, 64, tid, 256);
        stage_tile(Bu, 72, Wu + k0, 256, 128, 64, tid, 256);
        stage_tile(Bg, 72, Wg + k0, 256, 128, 64, tid, 256);
        wait_async();
        __syncthreads();
#pragma unroll
        for (int kk = 0; kk < 64; kk += 32) {
            v16bf a = frag_a(As + rt * 16 * 72 + kk, 72, lane);
#pragma unroll
            for (int j = 0; j < 4; ++j) {
                v16bf b0 = frag_b(Bu + (cg * 64 + j * 16) * 72 + kk, 72, lane);
                v16bf b1 = frag_b(Bg + (cg * 64 + j * 16) * 72 + kk, 72, lane);
                accU[j] = wmma_bf16(a, b0, accU[j]);
                accG[j] = wmma_bf16(a, b1, accG[j]);
            }
        }
    }
    const int rbase = row0 + rt * 16 + ((lane >> 4) << 3);
#pragma unroll
    for (int j = 0; j < 4; ++j) {
        int d = cg * 64 + j * 16 + (lane & 15);
        float bud = bu[d], bgd = bg[d];
#pragma unroll
        for (int r = 0; r < 8; ++r) {
            int token = rbase + r;
            int b = token >> 11;          // / (H*M)
            int h = (token >> 8) & 7;
            int m = token & 255;
            float f = tanhf(fmaxf(accU[j][r] + bud, 0.f));
            float g = 1.f / (1.f + expf(-(accG[j][r] + bgd)));
            long xi = ((long)b * M_ + m) * E_ + h * HD_ + d;
            xnew[xi] = g * f + (1.f - g) * xold[xi];
        }
    }
}

// ---------------- layer norm over 1024, regenerating bf16 layouts ----------------
__global__ __launch_bounds__(256)
void layernorm_kernel(const float* __restrict__ src, const float* __restrict__ w,
                      const float* __restrict__ bb, float* __restrict__ xout,
                      bf16* __restrict__ xbout, bf16* __restrict__ xTout) {
    __shared__ float red[16];
    const long t = blockIdx.x;
    const int tid = threadIdx.x, lane = tid & 31, wave = tid >> 5;
    float4 v = *(const float4*)(src + t * E_ + tid * 4);
    float s  = v.x + v.y + v.z + v.w;
    float s2 = v.x * v.x + v.y * v.y + v.z * v.z + v.w * v.w;
#pragma unroll
    for (int m = 16; m; m >>= 1) { s += __shfl_xor(s, m, 32); s2 += __shfl_xor(s2, m, 32); }
    if (lane == 0) { red[wave] = s; red[8 + wave] = s2; }
    __syncthreads();
    if (tid == 0) {
        float S = 0.f, S2 = 0.f;
        for (int i = 0; i < 8; ++i) { S += red[i]; S2 += red[8 + i]; }
        float mu = S * (1.f / E_);
        red[0] = mu;
        red[1] = rsqrtf(S2 * (1.f / E_) - mu * mu + 1e-5f);
    }
    __syncthreads();
    float mu = red[0], rs = red[1];
    long b = t >> 8;
    int  m = (int)(t & 255);
    float vv[4] = { v.x, v.y, v.z, v.w };
#pragma unroll
    for (int i = 0; i < 4; ++i) {
        int e = tid * 4 + i;
        float y = (vv[i] - mu) * rs * w[e] + bb[e];
        xout[t * E_ + e] = y;
        xbout[t * E_ + e] = (bf16)y;
        if (xTout) {
            int h = e >> 7, d = e & 127;
            xTout[(((long)(b * H_ + h)) * HD_ + d) * M_ + m] = (bf16)y;
        }
    }
}

// ---------------- masked mean pool into feats[b][slot*E + e] ----------------
__global__ __launch_bounds__(256)
void pool_kernel(const float* __restrict__ src, const float* __restrict__ mask,
                 float* __restrict__ feats, bf16* __restrict__ featsb, int slot) {
    int b = blockIdx.x >> 2;
    int e = ((blockIdx.x & 3) << 8) + threadIdx.x;
    float s = 0.f, ms = 0.f;
    for (int m = 0; m < M_; ++m) {
        float mk = mask[b * M_ + m];
        s  += src[((long)b * M_ + m) * E_ + e] * mk;
        ms += mk;
    }
    float v = s / ms;
    long idx = (long)b * ((L_ + 1) * E_) + slot * E_ + e;
    feats[idx]  = v;
    featsb[idx] = (bf16)v;
}

// ---------------- weight transpose f32 [batch][In][Out] -> bf16 [batch][Out][In] ----------------
__global__ __launch_bounds__(256)
void convw_kernel(const float* __restrict__ W, bf16* __restrict__ Wt,
                  int In, int Out, long total) {
    long i = (long)blockIdx.x * 256 + threadIdx.x;
    if (i >= total) return;
    long per = (long)In * Out;
    long l = i / per, r = i - l * per;
    long ii = r / Out, o = r - ii * Out;
    Wt[l * per + o * In + ii] = (bf16)W[i];
}

// ---------------- initial derive: x copy + bf16 + head-transposed bf16 ----------------
__global__ __launch_bounds__(256)
void derive0_kernel(const float* __restrict__ att, float* __restrict__ x,
                    bf16* __restrict__ xb, bf16* __restrict__ xT) {
    long i = (long)blockIdx.x * 256 + threadIdx.x;
    float v = att[i];
    x[i]  = v;
    xb[i] = (bf16)v;
    int  e  = (int)(i & 1023);
    long bm = i >> 10;
    int  m  = (int)(bm & 255);
    long b  = bm >> 8;
    int h = e >> 7, d = e & 127;
    xT[(((long)(b * H_ + h)) * HD_ + d) * M_ + m] = (bf16)v;
}

__global__ __launch_bounds__(256)
void copy_kernel(const float* __restrict__ src, float* __restrict__ dst, long n) {
    long i = (long)blockIdx.x * 256 + threadIdx.x;
    if (i < n) dst[i] = src[i];
}

// ---------------- host launch ----------------
extern "C" void kernel_launch(void* const* d_in, const int* in_sizes, int n_in,
                              void* d_out, int out_size, void* d_ws, size_t ws_size,
                              hipStream_t stream) {
    const float* att_feats = (const float*)d_in[1];
    const float* att_mask  = (const float*)d_in[2];
    const float* mem_W  = (const float*)d_in[3];
    const float* mem_b  = (const float*)d_in[4];
    const float* gn_w   = (const float*)d_in[5];
    const float* gn_b   = (const float*)d_in[6];
    const float* upd_W  = (const float*)d_in[7];
    const float* upd_b  = (const float*)d_in[8];
    const float* gate_W = (const float*)d_in[9];
    const float* gate_b = (const float*)d_in[10];
    const float* ln_w   = (const float*)d_in[11];
    const float* ln_b   = (const float*)d_in[12];
    const float* proj_W = (const float*)d_in[13];
    const float* proj_b = (const float*)d_in[14];
    const float* lnf_w  = (const float*)d_in[15];
    const float* lnf_b  = (const float*)d_in[16];
    float* out = (float*)d_out;

    // workspace carve
    char* w = (char*)d_ws;
    auto carve = [&](size_t bytes) -> void* {
        void* p = (void*)w;
        w += (bytes + 255) & ~(size_t)255;
        return p;
    };
    const size_t TE = (size_t)TOK * E_;
    float* x      = (float*)carve(TE * 4);
    float* xnew   = (float*)carve(TE * 4);
    float* mem    = (float*)carve(TE * 4);
    bf16*  xb     = (bf16*)carve(TE * 2);
    bf16*  xT     = (bf16*)carve(TE * 2);
    bf16*  memb   = (bf16*)carve(TE * 2);
    bf16*  inp    = (bf16*)carve((size_t)BHM * 256 * 2);
    bf16*  memWt  = (bf16*)carve((size_t)L_ * E_ * E_ * 2);
    bf16*  updWt  = (bf16*)carve((size_t)L_ * HD_ * 256 * 2);
    bf16*  gateWt = (bf16*)carve((size_t)L_ * HD_ * 256 * 2);
    bf16*  projWt = (bf16*)carve((size_t)E_ * 5 * E_ * 2);
    float* feats  = (float*)carve((size_t)B_ * 5 * E_ * 4);
    bf16*  featsb = (bf16*)carve((size_t)B_ * 5 * E_ * 2);
    float* gv     = (float*)carve((size_t)B_ * E_ * 4);

    // weight conversion/transposition
    {
        long t1 = (long)L_ * E_ * E_;
        convw_kernel<<<(t1 + 255) / 256, 256, 0, stream>>>(mem_W, memWt, E_, E_, t1);
        long t2 = (long)L_ * 256 * HD_;
        convw_kernel<<<(t2 + 255) / 256, 256, 0, stream>>>(upd_W, updWt, 256, HD_, t2);
        convw_kernel<<<(t2 + 255) / 256, 256, 0, stream>>>(gate_W, gateWt, 256, HD_, t2);
        long t3 = (long)5 * E_ * E_;
        convw_kernel<<<(t3 + 255) / 256, 256, 0, stream>>>(proj_W, projWt, 5 * E_, E_, t3);
    }

    derive0_kernel<<<TE / 256, 256, 0, stream>>>(att_feats, x, xb, xT);
    pool_kernel<<<B_ * 4, 256, 0, stream>>>(x, att_mask, feats, featsb, 0);

    for (int l = 0; l < L_; ++l) {
        // mem = celu(x @ mem_W + mem_b)  (flop-dominant GEMM, 128x128 tiles, async double-buffer)
        gemm128_kernel<<<dim3(TOK / 128, E_ / 128), 256, G128_SMEM, stream>>>(
            xb, E_, memWt + (size_t)l * E_ * E_, E_, mem_b + l * E_, mem, E_, E_, 1);
        // group norm -> memb (bf16 keys)
        groupnorm_kernel<<<TOK * H_ / 8, 256, 0, stream>>>(mem, gn_w + l * E_, gn_b + l * E_, memb);
        // fused attention -> inp = [xh | ctx] bf16
        attn_kernel<<<dim3(B_ * H_, M_ / 64), 256, ATTN_SMEM, stream>>>(xb, memb, xT, att_mask, inp);
        // dual GEMM + gating -> xnew
        updgate_kernel<<<BHM / 64, 256, 0, stream>>>(
            inp, updWt + (size_t)l * HD_ * 256, gateWt + (size_t)l * HD_ * 256,
            upd_b + l * HD_, gate_b + l * HD_, x, xnew);
        // layer norm -> x, xb, xT for next layer
        layernorm_kernel<<<TOK, 256, 0, stream>>>(xnew, ln_w + l * E_, ln_b + l * E_, x, xb, xT);
        pool_kernel<<<B_ * 4, 256, 0, stream>>>(x, att_mask, feats, featsb, l + 1);
    }

    // final projection: [64 x 5120] @ [5120 x 1024] + b
    gemm_bf16_kernel<<<dim3(1, E_ / 128), 256, 0, stream>>>(
        featsb, 5 * E_, projWt, 5 * E_, proj_b, gv, E_, 5 * E_, 0);
    // final layer norm -> d_out[0 : 64*1024]   (bf16 side output goes to scratch memb)
    layernorm_kernel<<<B_, 256, 0, stream>>>(gv, lnf_w, lnf_b, out, memb, nullptr);
    // x output -> d_out[65536 : ]
    copy_kernel<<<TE / 256, 256, 0, stream>>>(x, out + (size_t)B_ * E_, (long)TE);
}
